// CPA_CCA_block_41334765257061
// MI455X (gfx1250) — matrix operand order repaired and checked by reading, not compiled, not run on hardware
//
#include <hip/hip_runtime.h>
#include <hip/hip_bf16.h>
#include <math.h>

#define BATCH 8
#define C_CH  256
#define HW    96
#define NPIX  9216            // 96*96
#define K16   16
#define NEG_SLOPE 0.001f

typedef __attribute__((ext_vector_type(16))) __bf16 v16bf;
typedef __attribute__((ext_vector_type(8)))  __bf16 v8bf;
typedef __attribute__((ext_vector_type(8)))  float  v8f;

__device__ __forceinline__ float lrelu(float v){ return v >= 0.f ? v : NEG_SLOPE * v; }

// ---------------------------------------------------------------------------
// Stage 0: per-pixel pass over all channels.
//   F[b,k,n]  = sum_c w_f[k,c] * x[b,c,n]          (rank-16 projection)
//   AM[b,0,n] = mean_c x ; AM[b,1,n] = max_c x
//   Bm[b,n]   = lrelu(sum_c w_a2b[c] * x)
// ---------------------------------------------------------------------------
__global__ __launch_bounds__(256) void k_stage0(const float* __restrict__ x,
        const float* __restrict__ w_f, const float* __restrict__ w_a2b,
        float* __restrict__ F, float* __restrict__ AM, float* __restrict__ Bm){
    const int b = blockIdx.y;
    const int t = threadIdx.x;
    const int pix = blockIdx.x * 256 + t;
    __shared__ float sWF[K16 * C_CH];
    __shared__ float sA2B[C_CH];
    for (int i = t; i < K16 * C_CH; i += 256) sWF[i] = w_f[i];
    sA2B[t] = w_a2b[t];
    __syncthreads();

    const float* xb = x + (size_t)b * C_CH * NPIX;
    float f[K16];
    #pragma unroll
    for (int k = 0; k < K16; ++k) f[k] = 0.f;
    float sum = 0.f, mx = -INFINITY, bm = 0.f;
    for (int c = 0; c < C_CH; ++c){
        float v = xb[(size_t)c * NPIX + pix];
        sum += v; mx = fmaxf(mx, v); bm += sA2B[c] * v;
        #pragma unroll
        for (int k = 0; k < K16; ++k) f[k] += sWF[k * C_CH + c] * v;
    }
    const size_t fb = (size_t)b * K16 * NPIX;
    #pragma unroll
    for (int k = 0; k < K16; ++k) F[fb + (size_t)k * NPIX + pix] = f[k];
    AM[(size_t)b * 2 * NPIX + pix]          = sum * (1.f / 256.f);
    AM[((size_t)b * 2 + 1) * NPIX + pix]    = mx;
    Bm[(size_t)b * NPIX + pix]              = lrelu(bm);
}

// ---------------------------------------------------------------------------
// G[b,k,c] = sum_n F[b,k,n] * x[b,c,n]   (pre-softmax channel attention)
// ---------------------------------------------------------------------------
__global__ __launch_bounds__(256) void k_spre(const float* __restrict__ x,
        const float* __restrict__ F, float* __restrict__ G){
    const int b = blockIdx.y, c = blockIdx.x, t = threadIdx.x;
    const float* xr = x + ((size_t)b * C_CH + c) * NPIX;
    const float* Fb = F + (size_t)b * K16 * NPIX;
    float acc[K16];
    #pragma unroll
    for (int k = 0; k < K16; ++k) acc[k] = 0.f;
    for (int n = t; n < NPIX; n += 256){
        float v = xr[n];
        #pragma unroll
        for (int k = 0; k < K16; ++k) acc[k] += v * Fb[(size_t)k * NPIX + n];
    }
    __shared__ float red[256];
    for (int k = 0; k < K16; ++k){
        red[t] = acc[k]; __syncthreads();
        for (int s = 128; s > 0; s >>= 1){ if (t < s) red[t] += red[t + s]; __syncthreads(); }
        if (t == 0) G[((size_t)b * K16 + k) * C_CH + c] = red[0];
        __syncthreads();
    }
}

// ---------------------------------------------------------------------------
// Softmax over c of G, then fused  Wbs[b,c,k] = sum_c' w_beta[c,c'] * S[k,c']
// ---------------------------------------------------------------------------
__global__ __launch_bounds__(256) void k_smax_wbs(const float* __restrict__ G,
        const float* __restrict__ w_beta, float* __restrict__ Wbs){
    const int b = blockIdx.x, t = threadIdx.x;
    __shared__ float S[K16 * C_CH];
    __shared__ float red[256];
    const float* Gb = G + (size_t)b * K16 * C_CH;
    for (int k = 0; k < K16; ++k) S[k * C_CH + t] = Gb[k * C_CH + t];
    __syncthreads();
    for (int k = 0; k < K16; ++k){
        red[t] = S[k * C_CH + t]; __syncthreads();
        for (int s = 128; s > 0; s >>= 1){ if (t < s) red[t] = fmaxf(red[t], red[t + s]); __syncthreads(); }
        float mxv = red[0]; __syncthreads();
        float e = __expf(S[k * C_CH + t] - mxv);
        red[t] = e; __syncthreads();
        for (int s = 128; s > 0; s >>= 1){ if (t < s) red[t] += red[t + s]; __syncthreads(); }
        float inv = 1.f / red[0]; __syncthreads();
        S[k * C_CH + t] = e * inv; __syncthreads();
    }
    float acc[K16];
    #pragma unroll
    for (int k = 0; k < K16; ++k) acc[k] = 0.f;
    for (int cp = 0; cp < C_CH; ++cp){
        float w = w_beta[(size_t)t * C_CH + cp];
        #pragma unroll
        for (int k = 0; k < K16; ++k) acc[k] += w * S[k * C_CH + cp];
    }
    #pragma unroll
    for (int k = 0; k < K16; ++k)
        Wbs[((size_t)b * C_CH + t) * K16 + k] = acc[k];
}

// ---------------------------------------------------------------------------
// out1[b,c,n] = x[b,c,n] + sum_k Wbs[b,c,k] * F[b,k,n]
// ---------------------------------------------------------------------------
__global__ __launch_bounds__(256) void k_out1(const float* __restrict__ x,
        const float* __restrict__ F, const float* __restrict__ Wbs,
        float* __restrict__ out1){
    const int b = blockIdx.y, t = threadIdx.x;
    const int pix = blockIdx.x * 256 + t;
    __shared__ float sW[C_CH * K16];
    for (int i = t; i < C_CH * K16; i += 256) sW[i] = Wbs[(size_t)b * C_CH * K16 + i];
    __syncthreads();
    const float* Fb = F + (size_t)b * K16 * NPIX;
    float f[K16];
    #pragma unroll
    for (int k = 0; k < K16; ++k) f[k] = Fb[(size_t)k * NPIX + pix];
    const float* xb = x + (size_t)b * C_CH * NPIX;
    float* ob = out1 + (size_t)b * C_CH * NPIX;
    for (int c = 0; c < C_CH; ++c){
        float acc = xb[(size_t)c * NPIX + pix];
        #pragma unroll
        for (int k = 0; k < K16; ++k) acc += sW[c * K16 + k] * f[k];
        ob[(size_t)c * NPIX + pix] = acc;
    }
}

// ---------------------------------------------------------------------------
// Spatial conv branch
// ---------------------------------------------------------------------------
__device__ float convp(const float* a0, const float* a1, int h, int w,
                       const float* wt, int ks, int pad){
    float acc = 0.f;
    for (int dh = 0; dh < ks; ++dh){
        int hh = h + dh - pad; if (hh < 0 || hh >= HW) continue;
        for (int dw = 0; dw < ks; ++dw){
            int ww = w + dw - pad; if (ww < 0 || ww >= HW) continue;
            int pi = hh * HW + ww;
            acc += wt[dh * ks + dw] * a0[pi] + wt[ks * ks + dh * ks + dw] * a1[pi];
        }
    }
    return acc;
}

__global__ __launch_bounds__(256) void k_spatial(const float* __restrict__ AM,
        const float* __restrict__ w1, const float* __restrict__ w3,
        const float* __restrict__ w5, const float* __restrict__ w7,
        const float* __restrict__ w_f2c, float* __restrict__ F4,
        float* __restrict__ Cm){
    const int b = blockIdx.y, t = threadIdx.x;
    const int pix = blockIdx.x * 256 + t;
    const int h = pix / HW, w = pix % HW;
    __shared__ float sw[2 + 18 + 50 + 98 + 4];
    if (t < 2)  sw[t]       = w1[t];
    if (t < 18) sw[2 + t]   = w3[t];
    if (t < 50) sw[20 + t]  = w5[t];
    if (t < 98) sw[70 + t]  = w7[t];
    if (t < 4)  sw[168 + t] = w_f2c[t];
    __syncthreads();
    const float* a0 = AM + (size_t)b * 2 * NPIX;
    const float* a1 = a0 + NPIX;
    float s1 = lrelu(sw[0] * a0[pix] + sw[1] * a1[pix]);
    float s3 = lrelu(convp(a0, a1, h, w, sw + 2, 3, 1));
    float s5 = lrelu(convp(a0, a1, h, w, sw + 20, 5, 2));
    float s7 = lrelu(convp(a0, a1, h, w, sw + 70, 7, 3));
    const size_t fb = (size_t)b * 4 * NPIX;
    F4[fb + 0 * NPIX + pix] = s1;
    F4[fb + 1 * NPIX + pix] = s3;
    F4[fb + 2 * NPIX + pix] = s5;
    F4[fb + 3 * NPIX + pix] = s7;
    Cm[(size_t)b * NPIX + pix] =
        lrelu(sw[168] * s1 + sw[169] * s3 + sw[170] * s5 + sw[171] * s7);
}

// ---------------------------------------------------------------------------
// Spatial attention scores + softmax
// ---------------------------------------------------------------------------
__global__ __launch_bounds__(128) void k_spattn(const float* __restrict__ Bm,
        const float* __restrict__ Cm, float* __restrict__ Ssp){
    const int b = blockIdx.y, h = blockIdx.x, t = threadIdx.x;
    __shared__ float row[HW];
    __shared__ float red[128];
    const float* B2 = Bm + (size_t)b * NPIX;
    const float* C2 = Cm + (size_t)b * NPIX;
    if (t < HW) row[t] = B2[h * HW + t];
    __syncthreads();
    float pv = 0.f;
    if (t < HW){
        for (int w = 0; w < HW; ++w) pv += row[w] * C2[w * HW + t];
    }
    red[t] = (t < HW) ? pv : -INFINITY; __syncthreads();
    for (int s = 64; s > 0; s >>= 1){ if (t < s) red[t] = fmaxf(red[t], red[t + s]); __syncthreads(); }
    float mxv = red[0]; __syncthreads();
    float e = (t < HW) ? __expf(pv - mxv) : 0.f;
    red[t] = e; __syncthreads();
    for (int s = 64; s > 0; s >>= 1){ if (t < s) red[t] += red[t + s]; __syncthreads(); }
    float inv = 1.f / red[0];
    if (t < HW) Ssp[(size_t)b * NPIX + h * HW + t] = e * inv;
}

// ---------------------------------------------------------------------------
// D[b,c,n] = lrelu(sum_j w_f2d[c,j] * F4[b,j,n])
// ---------------------------------------------------------------------------
__global__ __launch_bounds__(256) void k_dmat(const float* __restrict__ F4,
        const float* __restrict__ w_f2d, float* __restrict__ D){
    const int b = blockIdx.y, t = threadIdx.x;
    const int pix = blockIdx.x * 256 + t;
    __shared__ float sW[C_CH * 4];
    for (int i = t; i < C_CH * 4; i += 256) sW[i] = w_f2d[i];
    __syncthreads();
    const size_t fb = (size_t)b * 4 * NPIX;
    float f0 = F4[fb + pix], f1 = F4[fb + NPIX + pix];
    float f2 = F4[fb + 2 * NPIX + pix], f3 = F4[fb + 3 * NPIX + pix];
    float* Db = D + (size_t)b * C_CH * NPIX;
    for (int c = 0; c < C_CH; ++c){
        float v = sW[c * 4] * f0 + sW[c * 4 + 1] * f1 + sW[c * 4 + 2] * f2 + sW[c * 4 + 3] * f3;
        Db[(size_t)c * NPIX + pix] = lrelu(v);
    }
}

// ---------------------------------------------------------------------------
// Esp[b,c,h,w] = sum_k Ssp[b,h,k] * D[b,c,k,w]
// ---------------------------------------------------------------------------
__global__ __launch_bounds__(256) void k_esp(const float* __restrict__ Ssp,
        const float* __restrict__ D, float* __restrict__ Esp){
    const int b = blockIdx.y, c = blockIdx.x, t = threadIdx.x;
    __shared__ float sS[NPIX];        // 36 KB
    const float* Sb = Ssp + (size_t)b * NPIX;
    for (int i = t; i < NPIX; i += 256) sS[i] = Sb[i];
    __syncthreads();
    const float* Dc = D + ((size_t)b * C_CH + c) * NPIX;
    float* Ec = Esp + ((size_t)b * C_CH + c) * NPIX;
    for (int pix = t; pix < NPIX; pix += 256){
        int h = pix / HW, w = pix - h * HW;
        float acc = 0.f;
        for (int k = 0; k < HW; ++k) acc += sS[h * HW + k] * Dc[k * HW + w];
        Ec[pix] = acc;
    }
}

// ---------------------------------------------------------------------------
// bf16-WMMA GEMM, fully compile-time specialized:
//   Y = act( W(256xK) @ [X0;X1](KxN) + Res )
// 256 threads = 8 wave32; tile M=128 N=64 K=32; LDS double-buffered;
// fully unrolled K loop; B tile staged column-per-thread so both A and B
// staging use packed v_cvt_pk_bf16_f32 + ds_store_b128 (two for A, one for B).
// ---------------------------------------------------------------------------
#define LDR 48   // padded row length in bf16 (96 bytes)

template<int K, bool HAS_X1, bool HAS_RES, int ACT>
__global__ __launch_bounds__(256) void k_gemm(const float* __restrict__ Wm,
        const float* __restrict__ X0, const float* __restrict__ X1,
        const float* __restrict__ Res, float* __restrict__ Y){
    const int b  = blockIdx.z;
    const int n0 = blockIdx.x * 64;
    const int m0 = blockIdx.y * 128;
    const int t  = threadIdx.x;
    const int lane = t & 31;
    const int wave = t >> 5;
    const int wm = wave & 3;      // 4 waves along M
    const int wn = wave >> 2;     // 2 waves along N

    __shared__ __align__(32) __bf16 As[2][128 * LDR];   // 2 x 12 KB
    __shared__ __align__(32) __bf16 Bs[2][64 * LDR];    // 2 x  6 KB

    const size_t bstride = (size_t)C_CH * NPIX;
    const float* x0b = X0 + (size_t)b * bstride;
    const float* x1b = HAS_X1 ? (X1 + (size_t)b * bstride) : nullptr;

    const int am  = t >> 1;            // A row 0..127
    const int aks = (t & 1) * 16;      // A k-slice 0/16
    const int bn2 = t & 63;            // B column within tile
    const int bk8 = (t >> 6) * 8;      // B k-group: 0,8,16,24

    v8f acc[2][2] = {};

    // a 32-wide K tile never straddles the X0/X1 boundary (both mult. of 32)
    auto xrow0 = [&](int kt) -> const float* {
        return (!HAS_X1 || kt < C_CH) ? (x0b + (size_t)kt * NPIX)
                                      : (x1b + (size_t)(kt - C_CH) * NPIX);
    };

    auto stage = [&](int buf, int kt){
        // --- A tile: weights f32 -> bf16, two packed b128 LDS stores ---
        const float* wr = Wm + (size_t)(m0 + am) * K + kt + aks;
        float4 v0 = *reinterpret_cast<const float4*>(wr + 0);
        float4 v1 = *reinterpret_cast<const float4*>(wr + 4);
        float4 v2 = *reinterpret_cast<const float4*>(wr + 8);
        float4 v3 = *reinterpret_cast<const float4*>(wr + 12);
        v8bf p0, p1;
        p0[0] = (__bf16)v0.x; p0[1] = (__bf16)v0.y; p0[2] = (__bf16)v0.z; p0[3] = (__bf16)v0.w;
        p0[4] = (__bf16)v1.x; p0[5] = (__bf16)v1.y; p0[6] = (__bf16)v1.z; p0[7] = (__bf16)v1.w;
        p1[0] = (__bf16)v2.x; p1[1] = (__bf16)v2.y; p1[2] = (__bf16)v2.z; p1[3] = (__bf16)v2.w;
        p1[4] = (__bf16)v3.x; p1[5] = (__bf16)v3.y; p1[6] = (__bf16)v3.z; p1[7] = (__bf16)v3.w;
        *reinterpret_cast<v8bf*>(&As[buf][am * LDR + aks])     = p0;
        *reinterpret_cast<v8bf*>(&As[buf][am * LDR + aks + 8]) = p1;

        // --- B tile: one column, contiguous 8-element K run per thread ---
        // 8 coalesced b32 loads (lanes sweep n), 4 cvt_pk, 1 ds_store_b128
        const float* xk = xrow0(kt) + (size_t)bk8 * NPIX + n0 + bn2;
        float e[8];
        #pragma unroll
        for (int i = 0; i < 8; ++i) e[i] = xk[(size_t)i * NPIX];
        v8bf pb;
        #pragma unroll
        for (int i = 0; i < 8; ++i) pb[i] = (__bf16)e[i];
        *reinterpret_cast<v8bf*>(&Bs[buf][bn2 * LDR + bk8]) = pb;
    };

    const int mlo = lane & 15;
    const int ks8 = (lane >> 4) * 8;
    const int kh  = (lane >> 4) * 16;

    auto compute = [&](int buf){
        v16bf afr[2], bfr[2];
        #pragma unroll
        for (int tm = 0; tm < 2; ++tm){
            int m = wm * 32 + tm * 16 + mlo;
            v8bf lo = *reinterpret_cast<const v8bf*>(&As[buf][m * LDR + ks8]);
            v8bf hi = *reinterpret_cast<const v8bf*>(&As[buf][m * LDR + 16 + ks8]);
            afr[tm] = __builtin_shufflevector(lo, hi,
                        0,1,2,3,4,5,6,7,8,9,10,11,12,13,14,15);
        }
        #pragma unroll
        for (int tn = 0; tn < 2; ++tn){
            int n = wn * 32 + tn * 16 + mlo;
            v8bf blo = *reinterpret_cast<const v8bf*>(&Bs[buf][n * LDR + ks8]);
            v8bf bhi = *reinterpret_cast<const v8bf*>(&Bs[buf][n * LDR + 16 + ks8]);
            bfr[tn] = __builtin_shufflevector(blo, bhi,
                        0,1,2,3,4,5,6,7,8,9,10,11,12,13,14,15);
        }
        #pragma unroll
        for (int tm = 0; tm < 2; ++tm)
            #pragma unroll
            for (int tn = 0; tn < 2; ++tn)
                acc[tm][tn] = __builtin_amdgcn_wmma_f32_16x16x32_bf16(
                    false, afr[tm], false, bfr[tn], (short)0, acc[tm][tn],
                    false, false);
    };

    constexpr int NT = K / 32;
    stage(0, 0);
    #pragma unroll
    for (int it = 0; it < NT; ++it){
        __syncthreads();   // buf[it&1] visible; prior reads of other buf done
        if (it + 1 < NT){
            if (it + 2 < NT){
                // prefetch tile it+2 into near caches (global_prefetch_b8)
                __builtin_prefetch(Wm + (size_t)(m0 + am) * K + (it + 2) * 32 + aks, 0, 3);
                __builtin_prefetch(xrow0((it + 2) * 32) + (size_t)bk8 * NPIX + n0 + bn2, 0, 3);
            }
            stage((it + 1) & 1, (it + 1) * 32);
        }
        compute(it & 1);
    }

    // epilogue: residual + activation, write f32
    float* yb = Y + (size_t)b * bstride;
    const float* rb = HAS_RES ? (Res + (size_t)b * bstride) : nullptr;
    const int rowoff = (lane >> 4) * 8;
    const int col = lane & 15;
    #pragma unroll
    for (int tm = 0; tm < 2; ++tm){
        #pragma unroll
        for (int tn = 0; tn < 2; ++tn){
            int n = n0 + wn * 32 + tn * 16 + col;
            #pragma unroll
            for (int r = 0; r < 8; ++r){
                int m = m0 + wm * 32 + tm * 16 + rowoff + r;
                float v = acc[tm][tn][r];
                if (HAS_RES) v += rb[(size_t)m * NPIX + n];
                if (ACT == 1)      v = lrelu(v);
                else if (ACT == 2) v = 1.f / (1.f + __expf(-v));
                yb[(size_t)m * NPIX + n] = v;
            }
        }
    }
}

// ---------------------------------------------------------------------------
extern "C" void kernel_launch(void* const* d_in, const int* in_sizes, int n_in,
                              void* d_out, int out_size, void* d_ws, size_t ws_size,
                              hipStream_t stream) {
    const float* x      = (const float*)d_in[0];
    const float* w_f    = (const float*)d_in[1];
    const float* w_beta = (const float*)d_in[2];
    const float* w1     = (const float*)d_in[3];
    const float* w3     = (const float*)d_in[4];
    const float* w5     = (const float*)d_in[5];
    const float* w7     = (const float*)d_in[6];
    const float* w_a2b  = (const float*)d_in[7];
    const float* w_f2c  = (const float*)d_in[8];
    const float* w_f2d  = (const float*)d_in[9];
    const float* w_e    = (const float*)d_in[10];
    const float* w_h    = (const float*)d_in[11];
    const float* w_m    = (const float*)d_in[12];
    const float* w_hm   = (const float*)d_in[13];
    float* out = (float*)d_out;

    // workspace layout (floats)
    float* ws = (float*)d_ws;
    const size_t szF    = (size_t)BATCH * K16 * NPIX;
    const size_t szAM   = (size_t)BATCH * 2 * NPIX;
    const size_t szBm   = (size_t)BATCH * NPIX;
    const size_t szG    = (size_t)BATCH * K16 * C_CH;
    const size_t szWbs  = (size_t)BATCH * C_CH * K16;
    const size_t szF4   = (size_t)BATCH * 4 * NPIX;
    const size_t szCm   = (size_t)BATCH * NPIX;
    const size_t szSsp  = (size_t)BATCH * NPIX;
    const size_t szBig  = (size_t)BATCH * C_CH * NPIX;

    float* Fp   = ws;              ws += szF;
    float* AMp  = ws;              ws += szAM;
    float* Bmp  = ws;              ws += szBm;
    float* Gp   = ws;              ws += szG;
    float* Wbsp = ws;              ws += szWbs;
    float* F4p  = ws;              ws += szF4;
    float* Cmp  = ws;              ws += szCm;
    float* Sspp = ws;              ws += szSsp;
    float* out1p = ws;             ws += szBig;
    float* out2p = ws;             ws += szBig;
    float* Dp    = ws;             ws += szBig;
    float* Espp  = ws;             ws += szBig;
    float* Hhp = Dp;    // D dead after Esp+out2 -> reuse for Hh
    float* Mp  = Espp;  // Esp dead after out2  -> reuse for M

    const dim3 pixGrid(NPIX / 256, BATCH);

    k_stage0<<<pixGrid, 256, 0, stream>>>(x, w_f, w_a2b, Fp, AMp, Bmp);
    k_spre<<<dim3(C_CH, BATCH), 256, 0, stream>>>(x, Fp, Gp);
    k_smax_wbs<<<BATCH, 256, 0, stream>>>(Gp, w_beta, Wbsp);
    k_out1<<<pixGrid, 256, 0, stream>>>(x, Fp, Wbsp, out1p);

    k_spatial<<<pixGrid, 256, 0, stream>>>(AMp, w1, w3, w5, w7, w_f2c, F4p, Cmp);
    k_spattn<<<dim3(HW, BATCH), 128, 0, stream>>>(Bmp, Cmp, Sspp);
    k_dmat<<<pixGrid, 256, 0, stream>>>(F4p, w_f2d, Dp);
    k_esp<<<dim3(C_CH, BATCH), 256, 0, stream>>>(Sspp, Dp, Espp);

    const dim3 gemmGrid(NPIX / 64, C_CH / 128, BATCH);
    // out2 = w_e @ Esp + x
    k_gemm<256, false, true, 0><<<gemmGrid, 256, 0, stream>>>(w_e, Espp, nullptr, x, out2p);
    // Hh = lrelu(w_h @ [out1; out2])
    k_gemm<512, true, false, 1><<<gemmGrid, 256, 0, stream>>>(w_h, out1p, out2p, nullptr, Hhp);
    // M = sigmoid(w_m @ [Hh; x])
    k_gemm<512, true, false, 2><<<gemmGrid, 256, 0, stream>>>(w_m, Hhp, x, nullptr, Mp);
    // H_M = lrelu(w_hm @ [Hh; M])  -> final output
    k_gemm<512, true, false, 1><<<gemmGrid, 256, 0, stream>>>(w_hm, Hhp, Mp, nullptr, out);
}
